// GQA_987842478904
// MI455X (gfx1250) — matrix operand order repaired and checked by reading, not compiled, not run on hardware
//
#include <hip/hip_runtime.h>

// ---------------- problem constants ----------------
#define Sdim 2048
#define Ddim 2048
#define NH   32
#define NKV  8
#define HDm  64
#define GQ   4            // NH / NKV
#define NQKV 3072         // NH*HDm + 2*NKV*HDm

typedef __attribute__((ext_vector_type(16))) __bf16 v16bf;
typedef __attribute__((ext_vector_type(8)))  __bf16 v8bf;
typedef __attribute__((ext_vector_type(8)))  float  v8f;

// ======================================================================
// 1) fp32 -> bf16 straight convert
// ======================================================================
__global__ __launch_bounds__(256) void cvt_f32_bf16(const float* __restrict__ in,
                                                    __bf16* __restrict__ out, int n) {
    int i = blockIdx.x * 256 + threadIdx.x;
    if (i < n) out[i] = (__bf16)in[i];
}

// ======================================================================
// 2) LDS-tiled transpose + convert: out[C][R] (bf16) = in[R][C] (f32)^T
//    R, C multiples of 32.
// ======================================================================
__global__ __launch_bounds__(256) void transpose_cvt(const float* __restrict__ in,
                                                     __bf16* __restrict__ out,
                                                     int R, int C) {
    __shared__ float tile[32][33];
    int ntc = C >> 5;
    int bx = blockIdx.x % ntc;   // col tile of input
    int by = blockIdx.x / ntc;   // row tile of input
    int tx = threadIdx.x & 31;
    int ty = threadIdx.x >> 5;   // 0..7
#pragma unroll
    for (int i = 0; i < 32; i += 8)
        tile[ty + i][tx] = in[(size_t)(by * 32 + ty + i) * C + bx * 32 + tx];
    __syncthreads();
#pragma unroll
    for (int i = 0; i < 32; i += 8)
        out[(size_t)(bx * 32 + ty + i) * R + by * 32 + tx] = (__bf16)tile[tx][ty + i];
}

// ======================================================================
// 3) bf16 WMMA GEMM:  C[M][N] (f32) = A[M][K] (bf16, row-major)
//                               x Bt[N][K] (bf16, row-major = B^T)
//    wave tile 32x64, block (256 thr = 8 waves, 2x4) tile 64x256.
//    Fragment layouts used here (wave32):
//      A 16x32:  lane L -> M=L%16, halves e: K = (L/16)*16 + e
//      B 32x16:  lane L -> N=L%16, halves e: K = (L/16)*16 + e   (contiguous, as A; not ISA table order)
//      C 16x16:  lane L, vgpr v -> M = v + 8*(L/16), N = L%16
// ======================================================================
__global__ __launch_bounds__(256) void gemm_bf16(const __bf16* __restrict__ A,
                                                 const __bf16* __restrict__ Bt,
                                                 float* __restrict__ C,
                                                 int M, int N, int K) {
    int wave = threadIdx.x >> 5;
    int lane = threadIdx.x & 31;
    int lhalf = lane >> 4;       // 0/1
    int l16   = lane & 15;
    int wm = wave >> 2;          // 0..1
    int wn = wave & 3;           // 0..3
    int nbm = M >> 6;
    int m0 = (blockIdx.x % nbm) * 64 + wm * 32;
    int n0 = (blockIdx.x / nbm) * 256 + wn * 64;

    v8f c[2][4] = {};
    for (int k = 0; k < K; k += 32) {
        v16bf a[2];
#pragma unroll
        for (int i = 0; i < 2; ++i) {
            const __bf16* ap = A + (size_t)(m0 + i * 16 + l16) * K + k + lhalf * 16;
            v8bf lo = *(const v8bf*)ap;
            v8bf hi = *(const v8bf*)(ap + 8);
            __builtin_prefetch(ap + 64, 0, 3);            // next-next k-step cacheline
#pragma unroll
            for (int e = 0; e < 8; ++e) { a[i][e] = lo[e]; a[i][e + 8] = hi[e]; }
        }
        v16bf b[4];
#pragma unroll
        for (int j = 0; j < 4; ++j) {
            const __bf16* bp = Bt + (size_t)(n0 + j * 16 + l16) * K + k + lhalf * 16;
            b[j] = *(const v16bf*)bp;
            __builtin_prefetch(bp + 64, 0, 3);
        }
#pragma unroll
        for (int i = 0; i < 2; ++i)
#pragma unroll
            for (int j = 0; j < 4; ++j)
                c[i][j] = __builtin_amdgcn_wmma_f32_16x16x32_bf16(
                    false, a[i], false, b[j], (short)0, c[i][j], false, false);
    }
#pragma unroll
    for (int i = 0; i < 2; ++i)
#pragma unroll
        for (int j = 0; j < 4; ++j) {
            int row0 = m0 + i * 16 + lhalf * 8;
            int col  = n0 + j * 16 + l16;
#pragma unroll
            for (int v = 0; v < 8; ++v)
                C[(size_t)(row0 + v) * N + col] = c[i][j][v];
        }
}

// ======================================================================
// 4) per-head RMSNorm + RoPE for Q and K, pack to bf16.
//    One wave per (s, head-slot); slot<NH -> Q head, else K head.
//    Qb layout [NH][S][64], Kb layout [NKV][S][64].
// ======================================================================
__global__ __launch_bounds__(256) void qk_post(const float* __restrict__ QKVf,
                                               const float* __restrict__ cosT,
                                               const float* __restrict__ sinT,
                                               const float* __restrict__ qg,
                                               const float* __restrict__ kg,
                                               __bf16* __restrict__ Qb,
                                               __bf16* __restrict__ Kb) {
    int wid  = (blockIdx.x * 256 + threadIdx.x) >> 5;
    int lane = threadIdx.x & 31;
    int s = wid / (NH + NKV);
    int h = wid % (NH + NKV);
    bool isQ = h < NH;
    int col = isQ ? h * HDm : NH * HDm + (h - NH) * HDm;
    const float* row = QKVf + (size_t)s * NQKV + col;
    float x0 = row[lane], x1 = row[lane + 32];
    float ss = x0 * x0 + x1 * x1;
#pragma unroll
    for (int m = 1; m < 32; m <<= 1) ss += __shfl_xor(ss, m, 32);
    float r = rsqrtf(ss * (1.0f / 64.0f) + 1e-6f);
    const float* g = isQ ? qg : kg;
    x0 = x0 * r * g[lane];
    x1 = x1 * r * g[lane + 32];
    float c0 = cosT[s * HDm + lane], c1 = cosT[s * HDm + lane + 32];
    float s0 = sinT[s * HDm + lane], s1 = sinT[s * HDm + lane + 32];
    float o0 = x0 * c0 - x1 * s0;          // rope: [-t2, t1] * sin + t * cos
    float o1 = x1 * c1 + x0 * s1;
    __bf16* dst = isQ ? (Qb + ((size_t)h * Sdim + s) * HDm)
                      : (Kb + ((size_t)(h - NH) * Sdim + s) * HDm);
    dst[lane]      = (__bf16)o0;
    dst[lane + 32] = (__bf16)o1;
}

// V pack, transposed: Vt[kv][d][s] = QKVf[s][(NH+NKV)*64 + kv*64 + d]
__global__ __launch_bounds__(256) void v_pack(const float* __restrict__ QKVf,
                                              __bf16* __restrict__ Vt) {
    int idx = blockIdx.x * 256 + threadIdx.x;          // over NKV*64*S
    int s  = idx & (Sdim - 1);
    int t  = idx >> 11;
    int d  = t & 63;
    int kv = t >> 6;
    Vt[idx] = (__bf16)QKVf[(size_t)s * NQKV + (NH + NKV) * HDm + kv * HDm + d];
}

// ======================================================================
// 5) causal flash attention, one wave = 16 query rows of one head.
//    block = 4 waves = 64 consecutive query rows, same head.
//    Streams 32-key blocks:  QK^T (4 wmma) -> online softmax -> P via LDS
//    C->A relayout -> P*V (4 wmma) into 16x64 f32 accumulator.
// ======================================================================
__global__ __launch_bounds__(128) void attn_kernel(const __bf16* __restrict__ Qb,
                                                   const __bf16* __restrict__ Kb,
                                                   const __bf16* __restrict__ Vt,
                                                   __bf16* __restrict__ Ab) {
    __shared__ __align__(32) __bf16 Pbuf[4][16 * 32];
    int wave = threadIdx.x >> 5, lane = threadIdx.x & 31;
    int lhalf = lane >> 4, l16 = lane & 15;
    int nqb = Sdim / 64;
    int h    = blockIdx.x / nqb;
    int qblk = blockIdx.x % nqb;
    int q0 = qblk * 64 + wave * 16;
    int kv = h / GQ;
    const __bf16* Qh = Qb + (size_t)h  * Sdim * HDm;
    const __bf16* Kh = Kb + (size_t)kv * Sdim * HDm;
    const __bf16* Vh = Vt + (size_t)kv * HDm * Sdim;
    __bf16* P = Pbuf[wave];

    // Q A-fragments for feature halves [0,32) and [32,64): reused all blocks
    v16bf aq[2];
#pragma unroll
    for (int f = 0; f < 2; ++f) {
        const __bf16* qp = Qh + (size_t)(q0 + l16) * HDm + f * 32 + lhalf * 16;
        v8bf lo = *(const v8bf*)qp;
        v8bf hi = *(const v8bf*)(qp + 8);
#pragma unroll
        for (int e = 0; e < 8; ++e) { aq[f][e] = lo[e]; aq[f][e + 8] = hi[e]; }
    }

    v8f o[4] = {};                 // O tile 16x64 (4 N-tiles)
    float mrow[8], lrow[8];
#pragma unroll
    for (int v = 0; v < 8; ++v) { mrow[v] = -__builtin_inff(); lrow[v] = 0.f; }
    const float sc = 0.125f * 1.44269504088896340736f;   // 1/sqrt(64)*log2(e)

    int nkt = (q0 >> 4) + 1;       // # of 16-key tiles needed (causal)
    for (int kt0 = 0; kt0 < nkt; kt0 += 2) {
        // ---- scores for key tiles kt0, kt0+1 ----
        v8f sv[2];
#pragma unroll
        for (int t = 0; t < 2; ++t) {
            int key = (kt0 + t) * 16 + l16;
            v8f acc = {};
#pragma unroll
            for (int f = 0; f < 2; ++f) {
                const __bf16* kp = Kh + (size_t)key * HDm + f * 32 + lhalf * 16;
                v16bf bk = *(const v16bf*)kp;
                __builtin_prefetch(kp + 32 * HDm, 0, 3);     // next 32-key block
                acc = __builtin_amdgcn_wmma_f32_16x16x32_bf16(
                    false, aq[f], false, bk, (short)0, acc, false, false);
            }
            sv[t] = acc;
        }
        // ---- causal mask + online softmax (row = v + 8*lhalf) ----
#pragma unroll
        for (int v = 0; v < 8; ++v) {
            int q = q0 + v + 8 * lhalf;
#pragma unroll
            for (int t = 0; t < 2; ++t)
                if ((kt0 + t) * 16 + l16 > q) sv[t][v] = -__builtin_inff();
            float mx = fmaxf(sv[0][v], sv[1][v]);
#pragma unroll
            for (int msk = 1; msk < 16; msk <<= 1) mx = fmaxf(mx, __shfl_xor(mx, msk, 32));
            float mnew  = fmaxf(mrow[v], mx);
            float alpha = exp2f((mrow[v] - mnew) * sc);
            float psum = 0.f;
#pragma unroll
            for (int t = 0; t < 2; ++t) {
                float p = exp2f((sv[t][v] - mnew) * sc);
                sv[t][v] = p;
                psum += p;
            }
#pragma unroll
            for (int msk = 1; msk < 16; msk <<= 1) psum += __shfl_xor(psum, msk, 32);
            lrow[v] = lrow[v] * alpha + psum;
            mrow[v] = mnew;
#pragma unroll
            for (int j = 0; j < 4; ++j) o[j][v] *= alpha;
        }
        // ---- P: C-layout -> A-layout via per-wave LDS staging ----
#pragma unroll
        for (int t = 0; t < 2; ++t)
#pragma unroll
            for (int v = 0; v < 8; ++v)
                P[(v + 8 * lhalf) * 32 + t * 16 + l16] = (__bf16)sv[t][v];
        asm volatile("s_wait_dscnt 0x0" ::: "memory");
        v16bf ap;
        {
            const __bf16* pr = P + l16 * 32 + lhalf * 16;
            v8bf plo = *(const v8bf*)pr;
            v8bf phi = *(const v8bf*)(pr + 8);
#pragma unroll
            for (int e = 0; e < 8; ++e) { ap[e] = plo[e]; ap[e + 8] = phi[e]; }
        }
        // ---- O += P (16x32) x V (32x64) ----
        int kb = kt0 * 16;
#pragma unroll
        for (int j = 0; j < 4; ++j) {
            const __bf16* vp = Vh + (size_t)(j * 16 + l16) * Sdim + kb + lhalf * 16;
            v16bf bv = *(const v16bf*)vp;
            __builtin_prefetch(vp + 32, 0, 3);               // next 32-key block
            o[j] = __builtin_amdgcn_wmma_f32_16x16x32_bf16(
                false, ap, false, bv, (short)0, o[j], false, false);
        }
    }
    // ---- normalize, store attn out bf16, layout [S][NH*64] ----
#pragma unroll
    for (int j = 0; j < 4; ++j)
#pragma unroll
        for (int v = 0; v < 8; ++v) {
            int m = v + 8 * lhalf;
            float val = o[j][v] / lrow[v];
            Ab[(size_t)(q0 + m) * (NH * HDm) + h * HDm + j * 16 + l16] = (__bf16)val;
        }
}

// ======================================================================
extern "C" void kernel_launch(void* const* d_in, const int* in_sizes, int n_in,
                              void* d_out, int out_size, void* d_ws, size_t ws_size,
                              hipStream_t stream) {
    (void)in_sizes; (void)n_in; (void)out_size; (void)ws_size;
    const float* x    = (const float*)d_in[0];
    // d_in[1] = causal mask, implicit in the attention kernel
    const float* cosT = (const float*)d_in[2];
    const float* sinT = (const float*)d_in[3];
    const float* Wq   = (const float*)d_in[4];
    const float* Wk   = (const float*)d_in[5];
    const float* Wv   = (const float*)d_in[6];
    const float* Wo   = (const float*)d_in[7];
    const float* qg   = (const float*)d_in[8];
    const float* kg   = (const float*)d_in[9];

    char* ws = (char*)d_ws;
    __bf16* xb   = (__bf16*)(ws);                     //  8 MB  x bf16 [S][D]
    __bf16* Wcat = (__bf16*)(ws + (8u  << 20));       // 12 MB  [3072][2048] = [Wq^T;Wk^T;Wv^T]
    __bf16* Wot  = (__bf16*)(ws + (20u << 20));       //  8 MB  Wo^T, stored [2048][2048]
    float*  QKVf = (float*) (ws + (28u << 20));       // 24 MB  [S][3072] f32
    __bf16* Qb   = (__bf16*)(ws + (52u << 20));       //  8 MB  [NH][S][64]
    __bf16* Kb   = (__bf16*)(ws + (60u << 20));       //  2 MB  [NKV][S][64]
    __bf16* Vt   = (__bf16*)(ws + (62u << 20));       //  2 MB  [NKV][64][S]
    __bf16* Ab   = (__bf16*)(ws + (64u << 20));       //  8 MB  attn out bf16 [S][2048]

    // 1) convert activations
    cvt_f32_bf16<<<(Sdim * Ddim) / 256, 256, 0, stream>>>(x, xb, Sdim * Ddim);

    // 2) transpose-convert weights (Bt layout = contiguous-K fragment loads)
    transpose_cvt<<<(Ddim / 32) * (Ddim / 32), 256, 0, stream>>>(Wq, Wcat, Ddim, NH * HDm);
    transpose_cvt<<<(Ddim / 32) * ((NKV * HDm) / 32), 256, 0, stream>>>(Wk, Wcat + (size_t)NH * HDm * Ddim, Ddim, NKV * HDm);
    transpose_cvt<<<(Ddim / 32) * ((NKV * HDm) / 32), 256, 0, stream>>>(Wv, Wcat + (size_t)(NH + NKV) * HDm * Ddim, Ddim, NKV * HDm);
    transpose_cvt<<<(Ddim / 32) * (Ddim / 32), 256, 0, stream>>>(Wo, Wot, NH * HDm, Ddim);

    // 3) fused QKV projection GEMM: [2048 x 3072] = xb @ Wcat^T
    gemm_bf16<<<(Sdim / 64) * (NQKV / 256), 256, 0, stream>>>(xb, Wcat, QKVf, Sdim, NQKV, Ddim);

    // 4) RMSNorm + RoPE + pack
    qk_post<<<(Sdim * (NH + NKV)) / 8, 256, 0, stream>>>(QKVf, cosT, sinT, qg, kg, Qb, Kb);
    v_pack<<<(NKV * HDm * Sdim) / 256, 256, 0, stream>>>(QKVf, Vt);

    // 5) causal flash attention
    attn_kernel<<<NH * (Sdim / 64), 128, 0, stream>>>(Qb, Kb, Vt, Ab);

    // 6) output projection GEMM -> f32 out
    gemm_bf16<<<(Sdim / 64) * (Ddim / 256), 256, 0, stream>>>(Ab, Wot, (float*)d_out, Sdim, Ddim, Ddim);
}
